// Edge_block_6081673691618
// MI455X (gfx1250) — matrix operand order repaired and checked by reading, not compiled, not run on hardware
//
#include <hip/hip_runtime.h>
#include <hip/hip_bf16.h>
#include <math.h>

typedef _Float16 v16h __attribute__((ext_vector_type(16)));
typedef float    v8f  __attribute__((ext_vector_type(8)));
typedef unsigned int u32x4 __attribute__((ext_vector_type(4)));
typedef int          i32x8 __attribute__((ext_vector_type(8)));
typedef int          i32x4 __attribute__((ext_vector_type(4)));

#define HWIN 512
#define H    256
#define W    256
#define P    65536      // H*W
#define NB   2
#define EPS  1e-5f

static __device__ __forceinline__ int imin(int a, int b) { return a < b ? a : b; }
static __device__ __forceinline__ float clampf(float v, float lo, float hi) {
    return fminf(fmaxf(v, lo), hi);
}

// ---------------------------------------------------------------------------
// TDM: stage a contiguous row of `nelem` f32 from global into LDS.
// Descriptor per CDNA5 ISA (2-D tensor, groups
// 2/3 zero). Issued by a single lane; TDM ignores EXEC; completion tracked
// with TENSORcnt. This toolchain exposes the 6-arg builtin
// (g0, g1, g2, g3, g_extra, cpol).
// ---------------------------------------------------------------------------
#if __has_builtin(__builtin_amdgcn_tensor_load_to_lds) && \
    __has_builtin(__builtin_amdgcn_s_wait_tensorcnt)
#define HAVE_TDM 1
#else
#define HAVE_TDM 0
#endif

#if HAVE_TDM
static __device__ __forceinline__ void tdm_load_1d(const float* g, float* lds, int nelem) {
    unsigned long long ga = (unsigned long long)(size_t)g;
    unsigned laddr = (unsigned)(unsigned long long)(size_t)lds;  // low 32b = LDS offset
    u32x4 g0;
    g0[0] = 1u;                                        // count=1, user descriptor
    g0[1] = laddr;                                     // lds_addr [63:32]
    g0[2] = (unsigned)ga;                              // global_addr[31:0]
    g0[3] = ((unsigned)(ga >> 32) & 0x1FFFFFFu) | (2u << 30);  // ga[56:32] | type=2
    i32x8 g1;
    g1[0] = 0x20000;                                   // wg_mask=0, data_size=2 (4B)
    g1[1] = (nelem & 0xFFFF) << 16;                    // tensor_dim0[15:0] @ bits 63:48
    g1[2] = ((nelem >> 16) & 0xFFFF) | (1 << 16);      // tensor_dim0[31:16] | tensor_dim1=1
    g1[3] = (nelem & 0xFFFF) << 16;                    // tile_dim0 @ bits 127:112
    g1[4] = 1;                                         // tile_dim1=1, tile_dim2=0
    g1[5] = nelem;                                     // tensor_dim0_stride[31:0]
    g1[6] = 0;
    g1[7] = 0;
    i32x4 z4 = {0, 0, 0, 0};
    i32x8 z8 = {0, 0, 0, 0, 0, 0, 0, 0};
    __builtin_amdgcn_tensor_load_to_lds(g0, g1, z4, z4, z8, 0);
    __builtin_amdgcn_s_wait_tensorcnt(0);
}
#endif

static __device__ __forceinline__ void stage_weights(const float* g, float* lds, int nelem) {
#if HAVE_TDM
    if (threadIdx.x == 0) tdm_load_1d(g, lds, nelem);
#else
    for (int i = threadIdx.x; i < nelem; i += blockDim.x) lds[i] = g[i];
#endif
    __syncthreads();
}

// ---------------------------------------------------------------------------
// Kernel 1: conv1 (3->64, 3x3, stride2, pad1) + BN + ReLU  -> h in f16
// ---------------------------------------------------------------------------
__global__ void conv1_bn_relu(const float* __restrict__ x, const float* __restrict__ w,
                              const float* __restrict__ g, const float* __restrict__ bb,
                              const float* __restrict__ m, const float* __restrict__ v,
                              _Float16* __restrict__ h16) {
    int idx = blockIdx.x * blockDim.x + threadIdx.x;   // NB*64*P
    int p  = idx & (P - 1);
    int oc = (idx >> 16) & 63;
    int b  = idx >> 22;
    int oy = p >> 8, ox = p & 255;
    // gfx1250 global_prefetch_b8 of the input neighborhood we are about to read
    __builtin_prefetch(&x[((size_t)(b * 3) * HWIN + imin(oy * 2, HWIN - 1)) * HWIN + ox * 2], 0, 3);
    float acc = 0.f;
    #pragma unroll
    for (int ic = 0; ic < 3; ++ic) {
        const float* xb = x + ((size_t)(b * 3 + ic) * HWIN) * HWIN;
        const float* wb = w + (oc * 3 + ic) * 9;
        #pragma unroll
        for (int ky = 0; ky < 3; ++ky) {
            int iy = oy * 2 - 1 + ky;
            if ((unsigned)iy >= HWIN) continue;
            #pragma unroll
            for (int kx = 0; kx < 3; ++kx) {
                int ix = ox * 2 - 1 + kx;
                if ((unsigned)ix >= HWIN) continue;
                acc += xb[iy * HWIN + ix] * wb[ky * 3 + kx];
            }
        }
    }
    float scale = g[oc] * rsqrtf(v[oc] + EPS);
    float val = (acc - m[oc]) * scale + bb[oc];
    h16[idx] = (_Float16)fmaxf(val, 0.f);
}

// ---------------------------------------------------------------------------
// Kernel 2: offset conv (64->10, 3x3, pad1) as WMMA GEMM, M=16(pad), K=576
// Weights TDM-staged to LDS once per block; 18 WMMA steps of k=32.
// ---------------------------------------------------------------------------
__global__ void off_conv_wmma(const _Float16* __restrict__ h16,
                              const float* __restrict__ w,      // (10,64,3,3) = 5760
                              const float* __restrict__ bias,   // (10)
                              float* __restrict__ off) {        // (NB,10,P)
    __shared__ float wlds[5760];
    stage_weights(w, wlds, 5760);

    int wt   = (blockIdx.x * blockDim.x + threadIdx.x) >> 5;    // 0..8191
    int lane = threadIdx.x & 31;
    int nlo = lane & 15, hi = lane >> 4;
    int b   = wt >> 12;
    int p   = ((wt & 4095) << 4) + nlo;
    int py = p >> 8, px = p & 255;
    const _Float16* hb = h16 + (size_t)(b * 64) * P;
    v8f c = {};
    for (int kk = 0; kk < 18; ++kk) {
        int kbase = kk * 32;
        v16h a, bm;
        #pragma unroll
        for (int i = 0; i < 16; ++i) {                 // A: weights from LDS, row M = nlo
            int k = kbase + ((i >> 3) << 4) + (hi << 3) + (i & 7);
            float wv = (nlo < 10) ? wlds[nlo * 576 + k] : 0.f;
            a[i] = (_Float16)wv;
        }
        #pragma unroll
        for (int i = 0; i < 16; ++i) {                 // B: im2col gather of h16
            int k  = kbase + ((i >> 3) << 4) + (hi << 3) + (i & 7);            // k = ic*9 + tap
            int ic = k / 9;
            int t  = k - ic * 9;
            int dy = t / 3 - 1;
            int dx = t - (t / 3) * 3 - 1;
            int yy = py + dy, xx = px + dx;
            _Float16 val = (_Float16)0.f;
            if ((unsigned)yy < H && (unsigned)xx < W)
                val = hb[(size_t)ic * P + yy * W + xx];
            bm[i] = val;
        }
        c = __builtin_amdgcn_wmma_f32_16x16x32_f16(false, a, false, bm, (short)0, c, false, false);
    }
    #pragma unroll
    for (int r = 0; r < 8; ++r) {
        int mrow = r + (hi << 3);
        if (mrow < 10)
            off[(size_t)(b * 10 + mrow) * P + p] = c[r] + bias[mrow];
    }
}

// ---------------------------------------------------------------------------
// Kernel 3: group-norm statistics (one block per (batch, group))
// ---------------------------------------------------------------------------
__global__ void gn_stats(const float* __restrict__ data, int chans_per_group, int nchan,
                         float* __restrict__ mean, float* __restrict__ rvar) {
    int ngroups = nchan / chans_per_group;
    int b = blockIdx.x / ngroups;
    int g = blockIdx.x - b * ngroups;
    const float* base = data + (size_t)(b * nchan + g * chans_per_group) * P;
    int n = chans_per_group * P;
    float s = 0.f, ss = 0.f;
    for (int i = threadIdx.x; i < n; i += blockDim.x) {
        float v = base[i];
        s += v; ss += v * v;
    }
    __shared__ float sh_s[256], sh_ss[256];
    sh_s[threadIdx.x] = s; sh_ss[threadIdx.x] = ss;
    __syncthreads();
    for (int o = blockDim.x >> 1; o > 0; o >>= 1) {
        if ((int)threadIdx.x < o) {
            sh_s[threadIdx.x]  += sh_s[threadIdx.x + o];
            sh_ss[threadIdx.x] += sh_ss[threadIdx.x + o];
        }
        __syncthreads();
    }
    if (threadIdx.x == 0) {
        float mu  = sh_s[0] / (float)n;
        float var = sh_ss[0] / (float)n - mu * mu;
        mean[blockIdx.x] = mu;
        rvar[blockIdx.x] = rsqrtf(var + EPS);
    }
}

// ---------------------------------------------------------------------------
// Kernel 4: apply GN + tanh, cumulative snake offsets, build coordinate map
// ---------------------------------------------------------------------------
__global__ void coord_map_k(const float* __restrict__ off,
                            const float* __restrict__ mean, const float* __restrict__ rvar,
                            const float* __restrict__ gg, const float* __restrict__ gb,
                            int morph, float* __restrict__ cmap) {       // (NB,5,P)
    int idx = blockIdx.x * blockDim.x + threadIdx.x;   // NB*P
    int b = idx >> 16;
    int p = idx & (P - 1);
    int chan0 = morph ? 5 : 0;
    float t[5];
    #pragma unroll
    for (int k = 0; k < 5; ++k) {
        int ch = chan0 + k;
        int g  = ch >> 1;                               // groups of 2 channels
        float v = off[(size_t)(b * 10 + ch) * P + p];
        v = (v - mean[b * 5 + g]) * rvar[b * 5 + g];
        v = v * gg[ch] + gb[ch];
        t[k] = tanhf(v);
    }
    float cum[5];
    cum[0] = t[0] + t[1]; cum[1] = t[1]; cum[2] = 0.f;
    cum[3] = t[3];        cum[4] = t[3] + t[4];
    float base = morph ? (float)(p & 255) : (float)(p >> 8);
    #pragma unroll
    for (int k = 0; k < 5; ++k)
        cmap[(size_t)(b * 5 + k) * P + p] = clampf(base + cum[k], 0.f, 255.f);
}

// ---------------------------------------------------------------------------
// Kernel 5: fused bilinear sampling + snake conv as WMMA GEMM
// M=16 (8 valid out ch), K = 5 taps * 64 ch = 320 -> 10 WMMA steps
// feat never materialized; weights TDM-staged to LDS.
// ---------------------------------------------------------------------------
__global__ void dsc_wmma(const _Float16* __restrict__ h16,
                         const float* __restrict__ cmap,     // (NB,5,P)
                         const float* __restrict__ w,        // (8,64,5) flat = 2560
                         const float* __restrict__ bias,     // (8)
                         int morph,
                         float* __restrict__ out) {          // (NB,8,P)
    __shared__ float wlds[2560];
    stage_weights(w, wlds, 2560);

    int wt   = (blockIdx.x * blockDim.x + threadIdx.x) >> 5;
    int lane = threadIdx.x & 31;
    int nlo = lane & 15, hi = lane >> 4;
    int b = wt >> 12;
    int p = ((wt & 4095) << 4) + nlo;
    float fy = (float)(p >> 8), fx = (float)(p & 255);
    int   y0i[5], y1i[5], x0i[5], x1i[5];
    float wy[5], wx[5];
    #pragma unroll
    for (int r = 0; r < 5; ++r) {
        float cm = cmap[(size_t)(b * 5 + r) * P + p];
        float spread = (float)(r - 2);
        float yc = morph ? clampf(fy + spread, 0.f, 255.f) : cm;
        float xc = morph ? cm : clampf(fx + spread, 0.f, 255.f);
        float yf = floorf(yc), xf = floorf(xc);
        wy[r] = yc - yf; wx[r] = xc - xf;
        int y0 = (int)yf, x0 = (int)xf;
        y0i[r] = imin(y0, H - 1); y1i[r] = imin(y0 + 1, H - 1);
        x0i[r] = imin(x0, W - 1); x1i[r] = imin(x0 + 1, W - 1);
    }
    const _Float16* hb = h16 + (size_t)(b * 64) * P;
    v8f c = {};
    for (int kk = 0; kk < 10; ++kk) {
        int kbase = kk * 32;
        v16h a, bm;
        #pragma unroll
        for (int i = 0; i < 16; ++i) {                 // A: weights from LDS
            int k = kbase + ((i >> 3) << 4) + (hi << 3) + (i & 7);
            int r = k >> 6, ic = k & 63;               // k = r*64 + ic
            float wv = (nlo < 8) ? wlds[(nlo * 64 + ic) * 5 + r] : 0.f;
            a[i] = (_Float16)wv;
        }
        #pragma unroll
        for (int i = 0; i < 16; ++i) {                 // B: bilinear samples
            int k = kbase + ((i >> 3) << 4) + (hi << 3) + (i & 7);
            int r = k >> 6, ic = k & 63;
            const _Float16* hc = hb + (size_t)ic * P;
            float v00 = (float)hc[y0i[r] * W + x0i[r]];
            float v01 = (float)hc[y0i[r] * W + x1i[r]];
            float v10 = (float)hc[y1i[r] * W + x0i[r]];
            float v11 = (float)hc[y1i[r] * W + x1i[r]];
            float vv = (1.f - wy[r]) * ((1.f - wx[r]) * v00 + wx[r] * v01)
                     +         wy[r] * ((1.f - wx[r]) * v10 + wx[r] * v11);
            bm[i] = (_Float16)vv;
        }
        c = __builtin_amdgcn_wmma_f32_16x16x32_f16(false, a, false, bm, (short)0, c, false, false);
    }
    #pragma unroll
    for (int r = 0; r < 8; ++r) {
        int mrow = r + (hi << 3);
        if (mrow < 8)
            out[(size_t)(b * 8 + mrow) * P + p] = c[r] + bias[mrow];
    }
}

// ---------------------------------------------------------------------------
// Kernel 6: apply GN (groups=2) + ReLU, write into f16 concat buffer
// ---------------------------------------------------------------------------
__global__ void gn_relu_cat(const float* __restrict__ out_raw,     // (NB,8,P)
                            const float* __restrict__ mean, const float* __restrict__ rvar,
                            const float* __restrict__ gg, const float* __restrict__ gb,
                            int branch, _Float16* __restrict__ cat16) {  // (NB,16,P)
    int idx = blockIdx.x * blockDim.x + threadIdx.x;   // NB*8*P
    int p  = idx & (P - 1);
    int ch = (idx >> 16) & 7;
    int b  = idx >> 19;
    int g  = ch >> 2;
    float v = out_raw[idx];
    v = (v - mean[b * 2 + g]) * rvar[b * 2 + g] * gg[ch] + gb[ch];
    cat16[(size_t)(b * 16 + branch * 8 + ch) * P + p] = (_Float16)fmaxf(v, 0.f);
}

// ---------------------------------------------------------------------------
// Kernel 7: conv2 (1x1, 16->64) as WMMA GEMM (K=16 padded to 32) + BN + ReLU
// Block = 4 waves = 4 m-tiles of the same 16-pixel n-tile; weights in LDS.
// ---------------------------------------------------------------------------
__global__ void conv2_bn_relu_wmma(const _Float16* __restrict__ cat16,
                                   const float* __restrict__ w,      // (64,16) = 1024
                                   const float* __restrict__ g2, const float* __restrict__ b2,
                                   const float* __restrict__ m2, const float* __restrict__ v2,
                                   float* __restrict__ out) {        // (NB,64,P)
    __shared__ float wlds[1024];
    stage_weights(w, wlds, 1024);

    int ntile = blockIdx.x;                 // 0..8191
    int mt    = threadIdx.x >> 5;           // m-tile = wave id (0..3)
    int lane  = threadIdx.x & 31;
    int nlo = lane & 15, hi = lane >> 4;
    int b = ntile >> 12;
    int p = ((ntile & 4095) << 4) + nlo;
    v16h a, bm;
    #pragma unroll
    for (int i = 0; i < 16; ++i) {
        int k = ((i >> 3) << 4) + (hi << 3) + (i & 7);
        a[i] = (k < 16) ? (_Float16)wlds[(mt * 16 + nlo) * 16 + k] : (_Float16)0.f;
    }
    #pragma unroll
    for (int i = 0; i < 16; ++i) {
        int k = ((i >> 3) << 4) + (hi << 3) + (i & 7);
        bm[i] = (k < 16) ? cat16[(size_t)(b * 16 + k) * P + p] : (_Float16)0.f;
    }
    v8f c = {};
    c = __builtin_amdgcn_wmma_f32_16x16x32_f16(false, a, false, bm, (short)0, c, false, false);
    #pragma unroll
    for (int r = 0; r < 8; ++r) {
        int oc = mt * 16 + r + (hi << 3);
        float scale = g2[oc] * rsqrtf(v2[oc] + EPS);
        float val = (c[r] - m2[oc]) * scale + b2[oc];
        out[(size_t)(b * 64 + oc) * P + p] = fmaxf(val, 0.f);
    }
}

// ---------------------------------------------------------------------------
// Host launcher
// ---------------------------------------------------------------------------
extern "C" void kernel_launch(void* const* d_in, const int* in_sizes, int n_in,
                              void* d_out, int out_size, void* d_ws, size_t ws_size,
                              hipStream_t stream) {
    const float* x        = (const float*)d_in[0];
    const float* conv1_w  = (const float*)d_in[1];
    const float* bn1_g    = (const float*)d_in[2];
    const float* bn1_b    = (const float*)d_in[3];
    const float* bn1_m    = (const float*)d_in[4];
    const float* bn1_v    = (const float*)d_in[5];
    const float* offx_w   = (const float*)d_in[6];
    const float* offx_b   = (const float*)d_in[7];
    const float* gnoffx_g = (const float*)d_in[8];
    const float* gnoffx_b = (const float*)d_in[9];
    const float* dscx_w   = (const float*)d_in[10];
    const float* dscx_b   = (const float*)d_in[11];
    const float* gnx_g    = (const float*)d_in[12];
    const float* gnx_b    = (const float*)d_in[13];
    const float* offy_w   = (const float*)d_in[14];
    const float* offy_b   = (const float*)d_in[15];
    const float* gnoffy_g = (const float*)d_in[16];
    const float* gnoffy_b = (const float*)d_in[17];
    const float* dscy_w   = (const float*)d_in[18];
    const float* dscy_b   = (const float*)d_in[19];
    const float* gny_g    = (const float*)d_in[20];
    const float* gny_b    = (const float*)d_in[21];
    const float* conv2_w  = (const float*)d_in[22];
    const float* bn2_g    = (const float*)d_in[23];
    const float* bn2_b    = (const float*)d_in[24];
    const float* bn2_m    = (const float*)d_in[25];
    const float* bn2_v    = (const float*)d_in[26];

    char* ws = (char*)d_ws;
    _Float16* h16      = (_Float16*)(ws);                       // 16,777,216 B
    float*    off_raw  = (float*)(ws + 16777216);               //  5,242,880 B (reused)
    float*    cmap     = (float*)(ws + 22020096);               //  2,621,440 B (reused)
    float*    out_raw0 = (float*)(ws + 24641536);               //  4,194,304 B
    float*    out_raw1 = (float*)(ws + 28835840);               //  4,194,304 B
    _Float16* cat16    = (_Float16*)(ws + 33030144);            //  4,194,304 B
    float*    stats    = (float*)(ws + 37224448);               //  small
    float* off_mean = stats;        float* off_rvar = stats + 16;
    float* og_mean  = stats + 32;   float* og_rvar  = stats + 48;

    // Stage 1: conv1 + BN + ReLU -> h16
    conv1_bn_relu<<<32768, 256, 0, stream>>>(x, conv1_w, bn1_g, bn1_b, bn1_m, bn1_v, h16);

    // Stage 2: two snake-conv branches (sequential; scratch reused)
    for (int br = 0; br < 2; ++br) {
        const float* ow  = br ? offy_w   : offx_w;
        const float* ob  = br ? offy_b   : offx_b;
        const float* gog = br ? gnoffy_g : gnoffx_g;
        const float* gob = br ? gnoffy_b : gnoffx_b;
        const float* dw  = br ? dscy_w   : dscx_w;
        const float* db  = br ? dscy_b   : dscx_b;
        const float* gng = br ? gny_g    : gnx_g;
        const float* gnb = br ? gny_b    : gnx_b;
        float* oraw      = br ? out_raw1 : out_raw0;

        off_conv_wmma<<<2048, 128, 0, stream>>>(h16, ow, ob, off_raw);
        gn_stats<<<10, 256, 0, stream>>>(off_raw, 2, 10, off_mean, off_rvar);
        coord_map_k<<<512, 256, 0, stream>>>(off_raw, off_mean, off_rvar, gog, gob, br, cmap);
        dsc_wmma<<<2048, 128, 0, stream>>>(h16, cmap, dw, db, br, oraw);
        gn_stats<<<4, 256, 0, stream>>>(oraw, 4, 8, og_mean, og_rvar);
        gn_relu_cat<<<4096, 256, 0, stream>>>(oraw, og_mean, og_rvar, gng, gnb, br, cat16);
    }

    // Stage 3: conv2 (1x1) + BN + ReLU -> output
    conv2_bn_relu_wmma<<<8192, 128, 0, stream>>>(cat16, conv2_w, bn2_g, bn2_b, bn2_m, bn2_v,
                                                 (float*)d_out);
}
